// Encoder2_60765197304598
// MI455X (gfx1250) — compile-verified, matrix operand order checked by reading, not run on hardware
//
#include <hip/hip_runtime.h>
#include <hip/hip_bf16.h>

// GraphSAGE 2-layer on MI455X (gfx1250, wave32, WMMA, LDS-staged tiles).
// degree -> scatter1 -> GEMM1 (mean@Wl1 + b + x@Wr1, relu)
//        -> scatter2 -> GEMM2 (+ fused h@W_s projection).
// Weights pre-packed to bf16 in WMMA B-fragment order; A tiles staged in LDS
// as bf16 once per block (mean scaling fused into staging).

#define NNODES 50000
#define DHID   128
#define DS_OUT 64
#define SH_LD  136   // bf16 row stride: 128 + 8 pad (keeps 16B runs aligned)

typedef __attribute__((ext_vector_type(16))) __bf16 v16bf;
typedef __attribute__((ext_vector_type(8)))  __bf16 v8bf;
typedef __attribute__((ext_vector_type(8)))  float  v8f;

// CDNA5 16-bit A-matrix 16x32 per-lane K mapping (ISA 7.12.2):
// lanes 0-15 hold K 0..7 / 16..23, lanes 16-31 hold K 8..15 / 24..31.
__device__ __forceinline__ int k_local(int i, int half) {
    return i + (half << 3) + ((i >= 8) ? 8 : 0);
}

// Build an A fragment from an LDS bf16 tile: two contiguous 8-elem K runs
// per lane -> two ds_load_b128.
__device__ __forceinline__ v16bf lds_a_frag(const __bf16* sh, int mr, int k0, int half) {
    const __bf16* base = sh + mr * SH_LD + k0 + (half << 3);
    v8bf lo = *(const v8bf*)(base);        // K = k0+8h .. k0+8h+7
    v8bf hi = *(const v8bf*)(base + 16);   // K = k0+16+8h .. k0+16+8h+7
    return __builtin_shufflevector(lo, hi, 0, 1, 2, 3, 4, 5, 6, 7,
                                           8, 9, 10, 11, 12, 13, 14, 15);
}

__global__ void zero_f32(float* __restrict__ p, long long n) {
    long long i = (long long)blockIdx.x * blockDim.x + threadIdx.x;
    if (i < n) p[i] = 0.0f;
}

__global__ void degree_kernel(const int* __restrict__ dst,
                              float* __restrict__ cnt, int e) {
    int i = blockIdx.x * blockDim.x + threadIdx.x;
    if (i < e) atomicAdd(&cnt[dst[i]], 1.0f);
}

// One thread per (edge, 4-feature chunk): float4 gather + 4 f32 atomics.
// Whole 25.6 MB accumulator is resident in the 192 MB L2.
__global__ void scatter_kernel(const float* __restrict__ x,
                               const int* __restrict__ src,
                               const int* __restrict__ dst,
                               float* __restrict__ agg, int e) {
    int t = blockIdx.x * blockDim.x + threadIdx.x;
    int edge = t >> 5;
    if (edge >= e) return;
    int c = (t & 31) << 2;
    const float4 v = *(const float4*)(x + (size_t)src[edge] * DHID + c);
    float* a = agg + (size_t)dst[edge] * DHID + c;
    atomicAdd(a + 0, v.x);
    atomicAdd(a + 1, v.y);
    atomicAdd(a + 2, v.z);
    atomicAdd(a + 3, v.w);
}

// Pack row-major f32 W[k][n] into bf16 WMMA B-fragment order:
// out[((kb*nwt + w)*32 + lane)*16 + i] = W[(kb*32 + k_local(i, lane>>4))*ncols + w*16 + (lane&15)]
__global__ void pack_w_kernel(const float* __restrict__ W, int ncols, int nwt,
                              __bf16* __restrict__ out, int total) {
    int t = blockIdx.x * blockDim.x + threadIdx.x;
    if (t >= total) return;
    int i    = t & 15;
    int lane = (t >> 4) & 31;
    int rest = t >> 9;
    int w    = rest % nwt;
    int kb   = rest / nwt;
    int k    = kb * 32 + k_local(i, lane >> 4);
    int n    = w * 16 + (lane & 15);
    out[t] = (__bf16)W[(size_t)k * ncols + n];
}

// Block = 16 rows x 128 cols, 8 wave32 waves, each owns one 16x16 tile.
// acc = bias + mean@Wl + x@Wr with bf16 WMMA / f32 accumulate.
// Wsp != null (layer 2): fuse out_s = h @ W_s via the LDS-staged bf16 h tile.
__global__ void __launch_bounds__(256)
sage_gemm_kernel(const float* __restrict__ agg, const float* __restrict__ cnt,
                 const float* __restrict__ xin,
                 const __bf16* __restrict__ Wlp, const float* __restrict__ bl,
                 const __bf16* __restrict__ Wrp,
                 float* __restrict__ hout, int apply_relu,
                 const __bf16* __restrict__ Wsp, float* __restrict__ outs)
{
    __shared__ __bf16 shM[16 * SH_LD];   // mean tile (bf16, scale fused)
    __shared__ __bf16 shX[16 * SH_LD];   // root tile (bf16)

    const int tid  = threadIdx.x;
    const int lane = tid & 31;
    const int wave = tid >> 5;
    const int half = lane >> 4;
    const int mr   = lane & 15;
    const int m0   = blockIdx.x * 16;    // N % 16 == 0 (50000/16 = 3125)
    const int ncol = wave * 16 + mr;

    // ---- cooperative tile staging: 8 f32 elems per thread per matrix ----
    {
        const int srow = tid >> 4;               // 0..15
        const int scol = (tid & 15) << 3;        // 0,8,...,120
        const size_t rbase = (size_t)(m0 + srow) * DHID + scol;
        const float inv = 1.0f / fmaxf(cnt[m0 + srow], 1.0f);
        const float4 a0 = *(const float4*)(agg + rbase);
        const float4 a1 = *(const float4*)(agg + rbase + 4);
        const float4 x0 = *(const float4*)(xin + rbase);
        const float4 x1 = *(const float4*)(xin + rbase + 4);
        __bf16* dm = shM + srow * SH_LD + scol;
        __bf16* dx = shX + srow * SH_LD + scol;
        dm[0] = (__bf16)(a0.x * inv); dm[1] = (__bf16)(a0.y * inv);
        dm[2] = (__bf16)(a0.z * inv); dm[3] = (__bf16)(a0.w * inv);
        dm[4] = (__bf16)(a1.x * inv); dm[5] = (__bf16)(a1.y * inv);
        dm[6] = (__bf16)(a1.z * inv); dm[7] = (__bf16)(a1.w * inv);
        dx[0] = (__bf16)x0.x; dx[1] = (__bf16)x0.y;
        dx[2] = (__bf16)x0.z; dx[3] = (__bf16)x0.w;
        dx[4] = (__bf16)x1.x; dx[5] = (__bf16)x1.y;
        dx[6] = (__bf16)x1.z; dx[7] = (__bf16)x1.w;
    }
    __syncthreads();

    v8f acc;
#pragma unroll
    for (int r = 0; r < 8; ++r) acc[r] = bl[ncol];   // bias folded into C

#pragma unroll
    for (int kb = 0; kb < 4; ++kb) {
        const size_t foff = ((size_t)(kb * 8 + wave) * 32 + lane) * 16;
        if (kb < 3) {   // global_prefetch_b8 for next packed K-slab
            __builtin_prefetch(Wlp + foff + 256 * 16, 0, 3);
            __builtin_prefetch(Wrp + foff + 256 * 16, 0, 3);
        }
        v16bf bL = *(const v16bf*)(Wlp + foff);   // 32B packed fragment
        v16bf bR = *(const v16bf*)(Wrp + foff);
        v16bf am = lds_a_frag(shM, mr, kb * 32, half);
        v16bf ax = lds_a_frag(shX, mr, kb * 32, half);
        acc = __builtin_amdgcn_wmma_f32_16x16x32_bf16(
                  false, am, false, bL, (short)0, acc, false, false);
        acc = __builtin_amdgcn_wmma_f32_16x16x32_bf16(
                  false, ax, false, bR, (short)0, acc, false, false);
    }

#pragma unroll
    for (int r = 0; r < 8; ++r) {
        float v = acc[r];
        if (apply_relu) v = fmaxf(v, 0.0f);
        hout[(size_t)(m0 + r + (half << 3)) * DHID + ncol] = v;   // M = r + 8*half
        acc[r] = v;
    }

    if (Wsp != nullptr) {            // layer 2: fuse out_s = h @ W_s
        __syncthreads();             // everyone done reading shM
#pragma unroll
        for (int r = 0; r < 8; ++r)
            shM[(r + (half << 3)) * SH_LD + ncol] = (__bf16)acc[r];
        __syncthreads();
        if (wave < 4) {              // 64 out cols -> 4 waves of 16
            const int nc2 = wave * 16 + mr;
            v8f acc2 = {0.f, 0.f, 0.f, 0.f, 0.f, 0.f, 0.f, 0.f};
#pragma unroll
            for (int kb = 0; kb < 4; ++kb) {
                v16bf a2 = lds_a_frag(shM, mr, kb * 32, half);
                v16bf b2 = *(const v16bf*)(Wsp +
                              ((size_t)(kb * 4 + wave) * 32 + lane) * 16);
                acc2 = __builtin_amdgcn_wmma_f32_16x16x32_bf16(
                           false, a2, false, b2, (short)0, acc2, false, false);
            }
#pragma unroll
            for (int r = 0; r < 8; ++r)
                outs[(size_t)(m0 + r + (half << 3)) * DS_OUT + nc2] = acc2[r];
        }
    }
}

extern "C" void kernel_launch(void* const* d_in, const int* in_sizes, int n_in,
                              void* d_out, int out_size, void* d_ws, size_t ws_size,
                              hipStream_t stream) {
    (void)n_in; (void)out_size; (void)ws_size;

    const float* x   = (const float*)d_in[0];
    const int*   ei  = (const int*)d_in[1];     // edge_index [2, E]
    const float* Wl1 = (const float*)d_in[2];
    const float* bl1 = (const float*)d_in[3];
    const float* Wr1 = (const float*)d_in[4];
    const float* Wl2 = (const float*)d_in[5];
    const float* bl2 = (const float*)d_in[6];
    const float* Wr2 = (const float*)d_in[7];
    const float* Wsm = (const float*)d_in[8];

    const int E  = in_sizes[1] / 2;
    const int Nn = NNODES;
    const int* src = ei;
    const int* dst = ei + E;

    // Workspace: agg [N,128] | cnt [N] | h1 [N,128] | packed bf16 weights
    float* agg = (float*)d_ws;
    float* cnt = agg + (size_t)Nn * DHID;
    float* h1  = cnt + Nn;
    __bf16* Wl1p = (__bf16*)(h1 + (size_t)Nn * DHID);
    __bf16* Wr1p = Wl1p + 16384;
    __bf16* Wl2p = Wr1p + 16384;
    __bf16* Wr2p = Wl2p + 16384;
    __bf16* Wsp  = Wr2p + 16384;                 // 8192 elems

    // d_out: out_s [N,64] then h2 [N,128]
    float* outs = (float*)d_out;
    float* h2   = outs + (size_t)Nn * DS_OUT;

    const long long nAgg = (long long)Nn * DHID;
    const int scatterBlocks = (int)(((long long)E * 32 + 255) / 256);

    // Pack weights to WMMA-fragment-ordered bf16 (tiny, once per launch).
    pack_w_kernel<<<64, 256, 0, stream>>>(Wl1, DHID,   8, Wl1p, 16384);
    pack_w_kernel<<<64, 256, 0, stream>>>(Wr1, DHID,   8, Wr1p, 16384);
    pack_w_kernel<<<64, 256, 0, stream>>>(Wl2, DHID,   8, Wl2p, 16384);
    pack_w_kernel<<<64, 256, 0, stream>>>(Wr2, DHID,   8, Wr2p, 16384);
    pack_w_kernel<<<32, 256, 0, stream>>>(Wsm, DS_OUT, 4, Wsp,  8192);

    // Zero agg + cnt (contiguous), compute degrees once.
    zero_f32<<<(unsigned)((nAgg + Nn + 255) / 256), 256, 0, stream>>>(agg, nAgg + Nn);
    degree_kernel<<<(E + 255) / 256, 256, 0, stream>>>(dst, cnt, E);

    // ---- Layer 1 ----
    scatter_kernel<<<scatterBlocks, 256, 0, stream>>>(x, src, dst, agg, E);
    sage_gemm_kernel<<<Nn / 16, 256, 0, stream>>>(agg, cnt, x, Wl1p, bl1, Wr1p,
                                                  h1, /*relu=*/1, nullptr, nullptr);

    // ---- Layer 2 (+ fused W_s projection) ----
    zero_f32<<<(unsigned)((nAgg + 255) / 256), 256, 0, stream>>>(agg, nAgg);
    scatter_kernel<<<scatterBlocks, 256, 0, stream>>>(h1, src, dst, agg, E);
    sage_gemm_kernel<<<Nn / 16, 256, 0, stream>>>(agg, cnt, h1, Wl2p, bl2, Wr2p,
                                                  h2, /*relu=*/0, Wsp, outs);
}
